// AttentionLayer_71966472012577
// MI455X (gfx1250) — matrix operand order repaired and checked by reading, not compiled, not run on hardware
//
#include <hip/hip_runtime.h>
#include <hip/hip_bf16.h>

typedef __bf16 bf16_t;
typedef __attribute__((ext_vector_type(16))) __bf16 v16bf;
typedef __attribute__((ext_vector_type(8)))  __bf16 v8bf;
typedef __attribute__((ext_vector_type(8)))  float  v8f;

#define D_MODEL 512
#define SEQ     2048
#define BATCH   8
#define MTOT    (BATCH * SEQ)

// Attention kernel LDS layout (bytes):
//   [0, 128K)          : 16 x 2048 f32 score tile
//   [128K, 256K)       : 4 waves x 2 buffers x 16KB bf16 K-tile staging
//   [256K, 256K+512)   : 16 x 8 f32 reduction scratch
#define SMEM_SCORES_BYTES (16 * SEQ * 4)
#define SMEM_KSTAGE_BYTES (4 * 2 * 16 * D_MODEL * 2)
#define SMEM_TOTAL_BYTES  (SMEM_SCORES_BYTES + SMEM_KSTAGE_BYTES + 16 * 8 * 4)

// ---------------------------------------------------------------------------
// Kernel 1: transpose + convert Wq/Wk (f32 [F,D] row-major) -> bf16 W^T [D,F]
// so that B-fragment loads in the projection GEMM are contiguous per lane.
// ---------------------------------------------------------------------------
__global__ __launch_bounds__(256) void prep_weights(
    const float* __restrict__ Wq, const float* __restrict__ Wk,
    bf16_t* __restrict__ WqT, bf16_t* __restrict__ WkT) {
  int id = blockIdx.x * blockDim.x + threadIdx.x;      // 0 .. 2*512*512-1
  int n  = id >> 9;                                    // 0..1023
  int kk = id & 511;
  if (n < 512) {
    WqT[n * 512 + kk] = (bf16_t)Wq[kk * 512 + n];
  } else {
    int nn = n - 512;
    WkT[nn * 512 + kk] = (bf16_t)Wk[kk * 512 + nn];
  }
}

// ---------------------------------------------------------------------------
// Kernel 2: projection GEMM  Out[m,n] = bf16( sum_k x[m,k]*W[k,n] + bias[n] )
// One wave per 16x16 output tile; 16 x v_wmma_f32_16x16x32_bf16 over K=512.
// ---------------------------------------------------------------------------
__global__ __launch_bounds__(32) void proj_kernel(
    const float* __restrict__ x, const bf16_t* __restrict__ WT,
    const float* __restrict__ bias, bf16_t* __restrict__ Out) {
  const int lane = threadIdx.x & 31;
  const int r    = lane & 15;       // row (A) / column (B) within tile
  const int hi   = lane >> 4;       // lane half selects K sub-range
  const int mbase = blockIdx.x * 16;
  const int nbase = blockIdx.y * 16;

  const float*  xrow = x  + (size_t)(mbase + r) * D_MODEL;
  const bf16_t* wrow = WT + (size_t)(nbase + r) * D_MODEL;

  v8f acc = {};
  for (int kb = 0; kb < D_MODEL; kb += 32) {
    v16bf a;
#pragma unroll
    for (int i = 0; i < 8; ++i) {
      a[i]     = (bf16_t)xrow[kb + hi * 16 + i];       // K = kb + hi*16 + i
      a[8 + i] = (bf16_t)xrow[kb + hi * 16 + 8 + i];   // K = kb + hi*16 + 8 + i
    }
    v16bf bfr = *(const v16bf*)(wrow + kb + hi * 16);  // K = kb + hi*16 .. +15
    acc = __builtin_amdgcn_wmma_f32_16x16x32_bf16(
        false, a, false, bfr, (short)0, acc, false, false);
  }

  const int   ncol = nbase + r;
  const float bv   = bias[ncol];
#pragma unroll
  for (int j = 0; j < 8; ++j) {
    const int row = mbase + j + 8 * hi;                // C layout: M = j + 8*(lane/16)
    Out[(size_t)row * D_MODEL + ncol] = (bf16_t)(acc[j] + bv);
  }
}

// ---------------------------------------------------------------------------
// Kernel 3: scores + softmax.  4 waves per (batch, 16 query rows).  K tiles
// are double-buffered into LDS with gfx1250 async global->LDS copies
// (ASYNCcnt), WMMAs read B fragments from LDS, scores land in a 128 KB LDS
// tile, then a cooperative row softmax streams alpha to HBM once.
// ---------------------------------------------------------------------------
__global__ __launch_bounds__(128) void attn_kernel(
    const bf16_t* __restrict__ Q, const bf16_t* __restrict__ K,
    float* __restrict__ out) {
  extern __shared__ char smem_raw[];
  float*  s   = (float*)smem_raw;
  bf16_t* kst = (bf16_t*)(smem_raw + SMEM_SCORES_BYTES);
  float*  red = (float*)(smem_raw + SMEM_SCORES_BYTES + SMEM_KSTAGE_BYTES);

  const int tid  = threadIdx.x;
  const int wave = tid >> 5;
  const int lane = tid & 31;
  const int r    = lane & 15;
  const int hi   = lane >> 4;
  const int b    = blockIdx.y;
  const int mblk = blockIdx.x;               // 0..127
  const int qrow0 = b * SEQ + mblk * 16;

  // Preload all 16 A fragments for this wave's 16 query rows (K=512).
  v16bf afrag[16];
  const bf16_t* qrow = Q + (size_t)(qrow0 + r) * D_MODEL;
#pragma unroll
  for (int kt = 0; kt < 16; ++kt) {
    const bf16_t* p = qrow + kt * 32;
    v8bf lo  = *(const v8bf*)(p + hi * 16);
    v8bf hi8 = *(const v8bf*)(p + hi * 16 + 8);
    afrag[kt] = __builtin_shufflevector(lo, hi8,
        0, 1, 2, 3, 4, 5, 6, 7, 8, 9, 10, 11, 12, 13, 14, 15);
  }

  // Async staging setup: each wave's key tile t is the contiguous 16 KB block
  // K[b*SEQ + (wave*32+t)*16 .. +16) rows.  32 x b128 per wave copies one tile.
  const unsigned kst_lds0 =
      (unsigned)(unsigned long long)(void*)kst + (unsigned)(wave * 2 * 16384);
  const unsigned long long kgbase =
      (unsigned long long)(const void*)(K + (size_t)b * SEQ * D_MODEL) +
      (unsigned long long)(wave * 32) * 16384ull;

#define STAGE_TILE(T, BUF)                                                     \
  do {                                                                         \
    unsigned _l0 = kst_lds0 + (unsigned)((BUF)*16384 + lane * 16);             \
    unsigned long long _g0 =                                                   \
        kgbase + (unsigned long long)(T)*16384ull + (unsigned)(lane * 16);     \
    _Pragma("unroll")                                                          \
    for (int _i = 0; _i < 32; ++_i) {                                          \
      unsigned _loff = _l0 + (unsigned)(_i * 512);                             \
      unsigned long long _ga = _g0 + (unsigned long long)(_i * 512);           \
      asm volatile("global_load_async_to_lds_b128 %0, %1, off" ::"v"(_loff),   \
                   "v"(_ga)                                                    \
                   : "memory");                                                \
    }                                                                          \
  } while (0)

  STAGE_TILE(0, 0);

  const float scale = 0.04419417382415922f;  // 1/sqrt(512)
  for (int t = 0; t < 32; ++t) {
    if (t + 1 < 32) {
      STAGE_TILE(t + 1, (t + 1) & 1);
      // <=32 outstanding => tile t's 32 copies (in-order) are complete
      asm volatile("s_wait_asynccnt 0x20" ::: "memory");
    } else {
      asm volatile("s_wait_asynccnt 0x0" ::: "memory");
    }

    const bf16_t* kbuf = kst + wave * 2 * 8192 + (t & 1) * 8192; // elements
    const int n0 = (wave * 32 + t) * 16;
    v8f acc = {};
#pragma unroll
    for (int kt = 0; kt < 16; ++kt) {
      // B fragment: column r = key (n0+r), K-range kt*32 + hi*16 .. +15
      v16bf bfr = *(const v16bf*)(kbuf + r * D_MODEL + kt * 32 + hi * 16);
      acc = __builtin_amdgcn_wmma_f32_16x16x32_bf16(
          false, afrag[kt], false, bfr, (short)0, acc, false, false);
    }
#pragma unroll
    for (int j = 0; j < 8; ++j)
      s[(j + 8 * hi) * SEQ + n0 + r] = acc[j] * scale;
  }
#undef STAGE_TILE
  __syncthreads();

  // Row softmax: 8 threads per row, 256 columns each.
  const int row = tid >> 3;                  // 0..15
  const int sub = tid & 7;
  float* srow = s + row * SEQ;
  const int nlo = sub * 256, nhi = nlo + 256;

  float m = -3.402823466e38f;
  for (int n = nlo; n < nhi; ++n) m = fmaxf(m, srow[n]);
  red[row * 8 + sub] = m;
  __syncthreads();
  float rm = red[row * 8];
#pragma unroll
  for (int i = 1; i < 8; ++i) rm = fmaxf(rm, red[row * 8 + i]);

  float lsum = 0.0f;
  for (int n = nlo; n < nhi; ++n) {
    float e = __expf(srow[n] - rm);
    srow[n] = e;
    lsum += e;
  }
  __syncthreads();                           // everyone done reading red (max)
  red[row * 8 + sub] = lsum;
  __syncthreads();
  float rs = 0.0f;
#pragma unroll
  for (int i = 0; i < 8; ++i) rs += red[row * 8 + i];
  const float inv = 1.0f / rs;

  float* orow = out + (size_t)b * SEQ * SEQ + (size_t)(mblk * 16 + row) * SEQ;
  for (int n = nlo; n < nhi; ++n) orow[n] = srow[n] * inv;
}

// ---------------------------------------------------------------------------
extern "C" void kernel_launch(void* const* d_in, const int* in_sizes, int n_in,
                              void* d_out, int out_size, void* d_ws, size_t ws_size,
                              hipStream_t stream) {
  (void)in_sizes; (void)n_in; (void)out_size; (void)ws_size;
  const float* x  = (const float*)d_in[0];
  const float* Wq = (const float*)d_in[1];
  const float* bq = (const float*)d_in[2];
  const float* Wk = (const float*)d_in[3];
  const float* bk = (const float*)d_in[4];
  // d_in[5]/d_in[6] (Wv, bv): v is unused in the reference's returned output.

  // Workspace layout (bf16 elements): WqT | WkT | Q | K  (~34.6 MB total)
  bf16_t* ws  = (bf16_t*)d_ws;
  bf16_t* WqT = ws;
  bf16_t* WkT = ws + 512 * 512;
  bf16_t* Qb  = ws + 2 * 512 * 512;
  bf16_t* Kb  = Qb + (size_t)MTOT * D_MODEL;

  prep_weights<<<2048, 256, 0, stream>>>(Wq, Wk, WqT, WkT);

  dim3 pg(MTOT / 16, D_MODEL / 16);
  proj_kernel<<<pg, 32, 0, stream>>>(x, WqT, bq, Qb);
  proj_kernel<<<pg, 32, 0, stream>>>(x, WkT, bk, Kb);

  size_t smem = SMEM_TOTAL_BYTES;            // 262656 B, fits 320 KB WGP LDS
  (void)hipFuncSetAttribute((const void*)attn_kernel,
                            hipFuncAttributeMaxDynamicSharedMemorySize,
                            (int)smem);
  dim3 ag(SEQ / 16, BATCH);
  attn_kernel<<<ag, 128, smem, stream>>>(Qb, Kb, (float*)d_out);
}
